// GNN_infer_51539607552378
// MI455X (gfx1250) — matrix operand order repaired and checked by reading, not compiled, not run on hardware
//
#include <hip/hip_runtime.h>
#include <hip/hip_bf16.h>
#include <math.h>

typedef __bf16 v16bf __attribute__((ext_vector_type(16)));
typedef __bf16 v8bf  __attribute__((ext_vector_type(8)));
typedef float  v8f   __attribute__((ext_vector_type(8)));

// Address-space-qualified pointer types (forces ds_* / global_* instead of flat_*)
typedef __attribute__((address_space(3))) __bf16        LBF;
typedef __attribute__((address_space(3))) float         LF32;
typedef __attribute__((address_space(1))) const __bf16  GCBF;
typedef __attribute__((address_space(1))) const float   GCF;
typedef __attribute__((address_space(1))) float         GF;

// ---------------------------------------------------------------------------
// Static layer tables (53 matmuls of the fused graph).
// ---------------------------------------------------------------------------
constexpr int NL = 53;
// 0 PCONV 1 HCONV 2 FCONV 3 BGCONV | 4-5 FULLHF | 6-7 FULLUP | 8 FULLATT | 9 FULLDEC
// 10-11 UPPH | 12-13 UPLH | 14-15 UPUP | 16-17 LOPH | 18-19 LOLH | 20-21 LOUP
// 22 UPATT 23 LOATT | 24 UPDEC 25 LODEC | 26-31 NODE1_j | 32-37 NODE2_j
// 38-43 UPD1_j | 44-49 UPD2_j | 50 PDCONV 51 HDCONV 52 FDCONV
constexpr int T_COUT[NL] = {60,20,10,10, 32,10, 32,10, 1, 20, 32,10, 32,10, 32,10,
                            32,10, 32,10, 32,10, 1,1, 40,20,
                            32,32,32,32,32,32, 10,10,10,10,10,10,
                            32,32,32,32,32,32, 10,10,10,10,10,10, 256,256,256};
constexpr int T_CIN[NL]  = {256,256,256,768, 20,32, 20,32, 10, 256, 40,32, 266,32, 20,32,
                            20,32, 266,32, 20,32, 10,10, 256,256,
                            306,306,306,306,306,306, 32,32,32,32,32,32,
                            20,20,20,20,20,20, 32,32,32,32,32,32, 70,30,20};
constexpr int T_MP[NL]   = {64,32,16,16, 32,16, 32,16, 16, 32, 32,16, 32,16, 32,16,
                            32,16, 32,16, 32,16, 16,16, 48,32,
                            32,32,32,32,32,32, 16,16,16,16,16,16,
                            32,32,32,32,32,32, 16,16,16,16,16,16, 256,256,256};
constexpr int T_CP[NL]   = {256,256,256,768, 64,32, 32,32, 32, 256, 64,32, 288,32, 32,32,
                            32,32, 288,32, 32,32, 32,32, 256,256,
                            320,320,320,320,320,320, 32,32,32,32,32,32,
                            32,32,32,32,32,32, 32,32,32,32,32,32, 96,32,32};

__host__ __device__ constexpr long woff(int L) {
  long o = 0; for (int i = 0; i < L; ++i) o += (long)T_MP[i] * T_CP[i]; return o;
}
__host__ __device__ constexpr long boff(int L) {
  long o = 0; for (int i = 0; i < L; ++i) o += T_MP[i]; return o;
}

// ---------------------------------------------------------------------------
// Geometry + LDS map
// ---------------------------------------------------------------------------
constexpr int  S_SP  = 96 * 96;
constexpr long TELEM = 4L * 256 * S_SP;
constexpr int  FSTR  = 264;   // feature tile stride: DW-stride ≡ 4 mod 8 -> conflict-free b128
constexpr int  SCH   = 600;   // state tile stride:   DW-stride ≡ 4 mod 8

constexpr int O_PT = 0;                 // p_fea tile [16][FSTR]
constexpr int O_HT = 16 * FSTR;
constexpr int O_FT = 32 * FSTR;
constexpr int O_ST = 48 * FSTR;         // state [16][SCH]
constexpr int LDS_ELEMS = 48 * FSTR + 16 * SCH;   // bf16 elems before att floats

// State channel map (all WMMA seg bases ≡ 0 mod 8 => 16B-aligned ds_load_b128)
constexpr int C_XPNEW = 0;    constexpr int C_BG1   = 60;
constexpr int C_XHU   = 96;   constexpr int C_XHL   = 106;  constexpr int C_BG2 = 116;
constexpr int C_XFNEW = 128;  constexpr int C_BG3   = 138;
constexpr int C_XP    = 160;  constexpr int C_XH    = 224;  constexpr int C_XHLC = 248;
constexpr int C_XF    = 264;  constexpr int C_XFMSG = 274;
constexpr int C_FDEC  = 296;  constexpr int C_UDEC  = 316;  constexpr int C_LDEC = 356;
constexpr int C_UPIN  = 376;  constexpr int C_LOIN  = 408;
constexpr int C_T32   = 440;  constexpr int C_TA    = 472;  constexpr int C_TB  = 482;
constexpr int C_NIN   = 496;  constexpr int C_UIN   = 560;
static_assert((C_XP & 7) == 0 && (C_XH & 7) == 0 && (C_XHLC & 7) == 0 && (C_XF & 7) == 0, "");
static_assert((C_UPIN & 7) == 0 && (C_LOIN & 7) == 0 && (C_T32 & 7) == 0, "");
static_assert((C_NIN & 7) == 0 && (C_UIN & 7) == 0 && (C_XHU & 7) == 0 && (C_XFNEW & 7) == 0, "");
static_assert(((C_XP + 40) & 7) == 0, "");

__device__ __forceinline__ __bf16 f2bf(float f) {
  unsigned u = __builtin_bit_cast(unsigned, f);
  unsigned r = (u + 0x7fffu + ((u >> 16) & 1u)) >> 16;
  return __builtin_bit_cast(__bf16, (unsigned short)r);
}
__device__ __forceinline__ float bf2f(__bf16 h) {
  unsigned u = ((unsigned)__builtin_bit_cast(unsigned short, h)) << 16;
  return __builtin_bit_cast(float, u);
}
__device__ __forceinline__ v8bf ldg8(GCBF* p) {
  return *(__attribute__((address_space(1))) const v8bf*)p;       // global_load_b128
}
__device__ __forceinline__ v8bf lds8(LBF* p) {
  return *(__attribute__((address_space(3))) const v8bf*)p;       // ds_load_b128
}
union VB { v16bf v; v8bf h[2]; };

// ---------------------------------------------------------------------------
// Weight packing: fold BN scale, f32->bf16, pad to WMMA-A layout.
// ---------------------------------------------------------------------------
struct PtrTab { const float* w[NL]; const float* s[NL]; const float* b[NL]; };

__global__ void pack_weights(PtrTab pt, __bf16* wb_, float* bb_) {
  GF* bbg = (GF*)bb_;
  __attribute__((address_space(1))) __bf16* wbg =
      (__attribute__((address_space(1))) __bf16*)wb_;
  const int L = blockIdx.x;
  const int cout = T_COUT[L], cin = T_CIN[L], mp = T_MP[L], cp = T_CP[L];
  int s1n = cin, s1p = cp, s2n = 0;
  if (L == 4)             { s1n = 10;  s1p = 32;  s2n = 10; }   // full_hf: [xh_u|pad|xh_l|pad]
  if (L == 12 || L == 18) { s1n = 256; s1p = 256; s2n = 10; }   // *_lh: [h_fea | xh_half|pad]
  const long wo = woff(L), bo = boff(L);
  GCF* w = (GCF*)pt.w[L];
  GCF* s = (GCF*)pt.s[L];
  GCF* b = (GCF*)pt.b[L];
  for (int e = threadIdx.x; e < mp * cp; e += blockDim.x) {
    const int m = e / cp, k = e % cp;
    int src = -1;
    if (k < s1p) { if (k < s1n) src = k; }
    else         { const int kk = k - s1p; if (kk < s2n) src = s1n + kk; }
    float v = 0.f;
    if (m < cout && src >= 0) { v = w[(long)m * cin + src]; if (pt.s[L]) v *= s[m]; }
    wbg[wo + e] = f2bf(v);
  }
  for (int m = threadIdx.x; m < mp; m += blockDim.x)
    bbg[bo + m] = (m < cout) ? b[m] : 0.f;
}

// ---------------------------------------------------------------------------
// Fused layer: relu( scale[pos]*(W @ act) + bias ) via v_wmma_f32_16x16x32_bf16.
// All loop bounds/offsets go through readfirstlane -> SGPR loops, EXEC stays
// all-ones around WMMA (ISA requirement), no exec-mask loop idioms.
// ---------------------------------------------------------------------------
__device__ __noinline__ void run_layer(
    GCBF* __restrict__ W, GCF* __restrict__ Bv, LBF* lds, LF32* attb,
    int cinPad, int coutPad, int coutReal,
    int o0, int st0, int kp0,
    int o1, int st1, int kp1,
    int o2, int st2, int kp2,
    int dstOff,                 // >=0: absolute LDS elem offset; <0: global dest
    GF* __restrict__ outG, size_t gBase,
    int scOff, int lane)
{
  cinPad   = __builtin_amdgcn_readfirstlane(cinPad);
  coutPad  = __builtin_amdgcn_readfirstlane(coutPad);
  coutReal = __builtin_amdgcn_readfirstlane(coutReal);
  o0 = __builtin_amdgcn_readfirstlane(o0); st0 = __builtin_amdgcn_readfirstlane(st0);
  kp0 = __builtin_amdgcn_readfirstlane(kp0);
  o1 = __builtin_amdgcn_readfirstlane(o1); st1 = __builtin_amdgcn_readfirstlane(st1);
  kp1 = __builtin_amdgcn_readfirstlane(kp1);
  o2 = __builtin_amdgcn_readfirstlane(o2); st2 = __builtin_amdgcn_readfirstlane(st2);
  kp2 = __builtin_amdgcn_readfirstlane(kp2);
  dstOff = __builtin_amdgcn_readfirstlane(dstOff);
  scOff  = __builtin_amdgcn_readfirstlane(scOff);

  const int ml = lane & 15, gr = lane >> 4;
  for (int mt = 0; mt < coutPad; mt += 16) {
    v8f acc = {0.f, 0.f, 0.f, 0.f, 0.f, 0.f, 0.f, 0.f};
    GCBF* arow = W + (size_t)(mt + ml) * cinPad + gr * 8;
    int kg = 0;
    // segment 0
    {
      LBF* bp = lds + o0 + ml * st0 + gr * 8;
      for (int kt = 0; kt < kp0; kt += 32) {
        VB a, b;
        a.h[0] = ldg8(arow + kg + kt);       a.h[1] = ldg8(arow + kg + kt + 16);
        b.h[0] = lds8(bp + kt);              b.h[1] = lds8(bp + kt + 16);
        acc = __builtin_amdgcn_wmma_f32_16x16x32_bf16(false, a.v, false, b.v,
                                                      (short)0, acc, false, false);
      }
      kg += kp0;
    }
    if (kp1) {
      LBF* bp = lds + o1 + ml * st1 + gr * 8;
      for (int kt = 0; kt < kp1; kt += 32) {
        VB a, b;
        a.h[0] = ldg8(arow + kg + kt);       a.h[1] = ldg8(arow + kg + kt + 16);
        b.h[0] = lds8(bp + kt);              b.h[1] = lds8(bp + kt + 16);
        acc = __builtin_amdgcn_wmma_f32_16x16x32_bf16(false, a.v, false, b.v,
                                                      (short)0, acc, false, false);
      }
      kg += kp1;
    }
    if (kp2) {
      LBF* bp = lds + o2 + ml * st2 + gr * 8;
      for (int kt = 0; kt < kp2; kt += 32) {
        VB a, b;
        a.h[0] = ldg8(arow + kg + kt);       a.h[1] = ldg8(arow + kg + kt + 16);
        b.h[0] = lds8(bp + kt);              b.h[1] = lds8(bp + kt + 16);
        acc = __builtin_amdgcn_wmma_f32_16x16x32_bf16(false, a.v, false, b.v,
                                                      (short)0, acc, false, false);
      }
    }
    const float s = (scOff >= 0) ? attb[scOff + ml] : 1.0f;
#pragma unroll
    for (int r = 0; r < 8; ++r) {
      const int m = mt + r + gr * 8;       // D layout: M = r + 8*(lane>>4)
      const float v = fmaxf(acc[r] * s + Bv[m], 0.0f);
      if (m < coutReal) {
        if (dstOff >= 0) lds[dstOff + ml * SCH + m] = f2bf(v);
        else             outG[gBase + (size_t)m * S_SP + ml] = v;
      }
    }
  }
}

// cout==1 sigmoid gate on VALU.
__device__ __noinline__ void run_att(GCBF* __restrict__ W, GCF* __restrict__ Bv,
                                     LBF* lds, int chanOff, LF32* o, int lane) {
  chanOff = __builtin_amdgcn_readfirstlane(chanOff);
  if (lane < 16) {
    float s = Bv[0];
    for (int c = 0; c < 10; ++c) s += bf2f(W[c]) * bf2f(lds[chanOff + lane * SCH + c]);
    o[lane] = 1.0f / (1.0f + __expf(-s));
  }
}

// ---------------------------------------------------------------------------
// Fully fused forward: 1 wave = 16 pixels, whole graph in LDS.
// ---------------------------------------------------------------------------
__global__ void __launch_bounds__(32) gnn_main(
    const __bf16* __restrict__ wb, const float* __restrict__ bb,
    const float* __restrict__ pf, const float* __restrict__ hf,
    const float* __restrict__ ff, float* __restrict__ out)
{
  extern __shared__ char smem[];
  LBF*  lds  = (LBF*)smem;
  LF32* attb = (LF32*)(lds + LDS_ELEMS);     // [attF 16 | attU 16 | attL 16]
  GCBF* wbG  = (GCBF*)wb;
  GCF*  bbG  = (GCF*)bb;
  GF*   outG = (GF*)out;

  const int lane = threadIdx.x & 31;
  const int ml = lane & 15, gr = lane >> 4;
  const int posBase = blockIdx.x * 16;
  const int bidx = posBase / S_SP;
  const int sp   = posBase % S_SP;
  const size_t inBase = (size_t)bidx * 256 * S_SP + sp;

  // ---- load + transpose features to position-major bf16 ----
  {
    GCF* srcs[3] = {(GCF*)pf, (GCF*)hf, (GCF*)ff};
    const int dofs[3] = {O_PT, O_HT, O_FT};
#pragma unroll
    for (int t = 0; t < 3; ++t) {
      GCF* s = srcs[t] + inBase + ml;
      const int d = dofs[t] + ml * FSTR;
      for (int c = gr; c < 256; c += 2) lds[d + c] = f2bf(s[(size_t)c * S_SP]);
    }
    typedef __attribute__((address_space(3))) unsigned long long LU64;
    for (int e = lane * 4; e < 16 * SCH; e += 128)  // zero state => padded-K reads safe
      *(LU64*)(lds + O_ST + e) = 0ull;
  }

#define RL(Lid, cp, mp, cr, o0,st0,kp0, o1,st1,kp1, o2,st2,kp2, dch, sc) \
  run_layer(wbG + woff(Lid), bbG + boff(Lid), lds, attb, cp, mp, cr,     \
            o0,st0,kp0, o1,st1,kp1, o2,st2,kp2, (dch), nullptr, 0, (sc), lane)
#define RG(Lid, cp, mp, cr, o0,kp0, gofs) \
  run_layer(wbG + woff(Lid), bbG + boff(Lid), lds, attb, cp, mp, cr,     \
            o0,SCH,kp0, 0,0,0, 0,0,0, -1, outG, (gofs), -1, lane)
#define SS(ch) (O_ST + (ch))

  // ---- encoders ----
  RL(0, 256, 64, 60, O_PT,FSTR,256, 0,0,0, 0,0,0, SS(C_XP),  -1);
  RL(1, 256, 32, 20, O_HT,FSTR,256, 0,0,0, 0,0,0, SS(C_XH),  -1);
  RL(2, 256, 16, 10, O_FT,FSTR,256, 0,0,0, 0,0,0, SS(C_XF),  -1);
  RL(3, 768, 16, 10, O_PT,FSTR,256, O_HT,FSTR,256, O_FT,FSTR,256, SS(C_BG1), -1);
  for (int e = lane; e < 160; e += 32) {           // bg copies + aligned xh_lower copy
    const int p = e / 10, c = e % 10;
    const __bf16 bg = lds[SS(C_BG1) + p * SCH + c];
    lds[SS(C_BG2)  + p * SCH + c] = bg;
    lds[SS(C_BG3)  + p * SCH + c] = bg;
    lds[SS(C_XHLC) + p * SCH + c] = lds[SS(C_XH) + p * SCH + 10 + c];
  }

  // ---- full graph ----
  RL(4, 64, 32, 32, SS(C_XH),SCH,32, SS(C_XHLC),SCH,32, 0,0,0, SS(C_T32),   -1);
  RL(5, 32, 16, 10, SS(C_T32),SCH,32, 0,0,0, 0,0,0,            SS(C_XFMSG), -1);
  RL(6, 32, 32, 32, SS(C_XF),SCH,32,  0,0,0, 0,0,0,            SS(C_T32),   -1);
  RL(7, 32, 16, 10, SS(C_T32),SCH,32, 0,0,0, 0,0,0,            SS(C_XFNEW), -1);
  run_att(wbG + woff(8),  bbG + boff(8),  lds, SS(C_XF),   attb + 0,  lane);
  run_att(wbG + woff(22), bbG + boff(22), lds, SS(C_XH),   attb + 16, lane);
  run_att(wbG + woff(23), bbG + boff(23), lds, SS(C_XHLC), attb + 32, lane);
  // decomp convs: W(x * a) == a * (W x)  -> per-pixel posScale
  RL(9,  256, 32, 20, O_HT,FSTR,256, 0,0,0, 0,0,0, SS(C_FDEC),  0);
  RL(24, 256, 48, 40, O_PT,FSTR,256, 0,0,0, 0,0,0, SS(C_UDEC), 16);
  RL(25, 256, 32, 20, O_PT,FSTR,256, 0,0,0, 0,0,0, SS(C_LDEC), 32);

  // ---- upper half ----
  RL(10, 64, 32, 32, SS(C_XP),SCH,64, 0,0,0, 0,0,0,              SS(C_T32), -1);
  RL(11, 32, 16, 10, SS(C_T32),SCH,32, 0,0,0, 0,0,0,             SS(C_TA),  -1);
  RL(12, 288, 32, 32, O_HT,FSTR,256, SS(C_XHLC),SCH,32, 0,0,0,   SS(C_T32), -1);
  RL(13, 32, 16, 10, SS(C_T32),SCH,32, 0,0,0, 0,0,0,             SS(C_TB),  -1);
  for (int e = lane; e < 160; e += 32) {
    const int p = e / 10, c = e % 10;
    lds[SS(C_UPIN) + p * SCH + c] = lds[SS(C_XH) + p * SCH + c];
    const float m = bf2f(lds[SS(C_TA) + p * SCH + c]) + bf2f(lds[SS(C_TB) + p * SCH + c])
                  + bf2f(lds[SS(C_FDEC) + p * SCH + c]);
    lds[SS(C_UPIN) + p * SCH + 10 + c] = f2bf(m);
  }
  RL(14, 32, 32, 32, SS(C_UPIN),SCH,32, 0,0,0, 0,0,0, SS(C_T32), -1);
  RL(15, 32, 16, 10, SS(C_T32),SCH,32,  0,0,0, 0,0,0, SS(C_XHU), -1);

  // ---- lower half ----
  RL(16, 32, 32, 32, SS(C_XP + 40),SCH,32, 0,0,0, 0,0,0,       SS(C_T32), -1);
  RL(17, 32, 16, 10, SS(C_T32),SCH,32, 0,0,0, 0,0,0,           SS(C_TA),  -1);
  RL(18, 288, 32, 32, O_HT,FSTR,256, SS(C_XH),SCH,32, 0,0,0,   SS(C_T32), -1);
  RL(19, 32, 16, 10, SS(C_T32),SCH,32, 0,0,0, 0,0,0,           SS(C_TB),  -1);
  for (int e = lane; e < 160; e += 32) {
    const int p = e / 10, c = e % 10;
    lds[SS(C_LOIN) + p * SCH + c] = lds[SS(C_XHLC) + p * SCH + c];
    const float m = bf2f(lds[SS(C_TA) + p * SCH + c]) + bf2f(lds[SS(C_TB) + p * SCH + c])
                  + bf2f(lds[SS(C_FDEC) + p * SCH + 10 + c]);
    lds[SS(C_LOIN) + p * SCH + 10 + c] = f2bf(m);
  }
  RL(20, 32, 32, 32, SS(C_LOIN),SCH,32, 0,0,0, 0,0,0, SS(C_T32), -1);
  RL(21, 32, 16, 10, SS(C_T32),SCH,32,  0,0,0, 0,0,0, SS(C_XHL), -1);

  // ---- part graph ----
  for (int j = 0; j < 6; ++j) {
    for (int e = lane; e < 800; e += 32) {         // stage 50-ch tail (parts != j)
      const int p = e / 50, c = e % 50;
      int pp = c / 10; const int cc = c - pp * 10; if (pp >= j) ++pp;
      lds[SS(C_NIN) + p * SCH + c] = lds[SS(C_XP) + p * SCH + pp * 10 + cc];
    }
    run_layer(wbG + woff(26 + j), bbG + boff(26 + j), lds, attb, 320, 32, 32,
              O_PT,FSTR,256, SS(C_NIN),SCH,64, 0,0,0, SS(C_T32), nullptr, 0, -1, lane);
    run_layer(wbG + woff(32 + j), bbG + boff(32 + j), lds, attb, 32, 16, 10,
              SS(C_T32),SCH,32, 0,0,0, 0,0,0, SS(C_TA), nullptr, 0, -1, lane);
    const int hchan = (j < 4) ? (C_UDEC + j * 10) : (C_LDEC + (j - 4) * 10);
    for (int e = lane; e < 160; e += 32) {
      const int p = e / 10, c = e % 10;
      lds[SS(C_UIN) + p * SCH + c] = lds[SS(C_XP) + p * SCH + j * 10 + c];
      const float m = bf2f(lds[SS(C_TA) + p * SCH + c]) + bf2f(lds[SS(hchan) + p * SCH + c]);
      lds[SS(C_UIN) + p * SCH + 10 + c] = f2bf(m);
    }
    run_layer(wbG + woff(38 + j), bbG + boff(38 + j), lds, attb, 32, 32, 32,
              SS(C_UIN),SCH,32, 0,0,0, 0,0,0, SS(C_T32), nullptr, 0, -1, lane);
    run_layer(wbG + woff(44 + j), bbG + boff(44 + j), lds, attb, 32, 16, 10,
              SS(C_T32),SCH,32, 0,0,0, 0,0,0, SS(C_XPNEW + j * 10), nullptr, 0, -1, lane);
  }

  // ---- decoders: stream f32 results straight to HBM ----
  const size_t gb = (size_t)bidx * 256 * S_SP + sp;
  RG(50, 96, 256, 256, SS(C_XPNEW), 96, gb);
  RG(51, 32, 256, 256, SS(C_XHU),   32, gb + TELEM);
  RG(52, 32, 256, 256, SS(C_XFNEW), 32, gb + 2 * TELEM);
#undef RL
#undef RG
#undef SS
}

// ---------------------------------------------------------------------------
// Host glue. d_in flattening: [p_fea, h_fea, f_fea] then params with sorted
// keys at every dict level (JAX pytree convention):
//   cbr:{b,s,w}  att:{b,w}  blk:{b1,b2,s1,s2,w1,w2}  node/upd stacked [6,...]
// ---------------------------------------------------------------------------
extern "C" void kernel_launch(void* const* d_in, const int* in_sizes, int n_in,
                              void* d_out, int out_size, void* d_ws, size_t ws_size,
                              hipStream_t stream) {
  (void)in_sizes; (void)out_size;
  if (n_in < 99) return;
  auto W = [&](int i) { return (const float*)d_in[i]; };

  PtrTab t{};
  t.w[0]=W(66); t.s[0]=W(65); t.b[0]=W(64);   // p_conv
  t.w[1]=W(31); t.s[1]=W(30); t.b[1]=W(29);   // h_conv
  t.w[2]=W(8);  t.s[2]=W(7);  t.b[2]=W(6);    // f_conv
  t.w[3]=W(5);  t.s[3]=W(4);  t.b[3]=W(3);    // bg_conv
  t.w[4]=W(21); t.s[4]=W(19); t.b[4]=W(17);   // full_hf
  t.w[5]=W(22); t.s[5]=W(20); t.b[5]=W(18);
  t.w[6]=W(27); t.s[6]=W(25); t.b[6]=W(23);   // full_update
  t.w[7]=W(28); t.s[7]=W(26); t.b[7]=W(24);
  t.w[8]=W(13); t.s[8]=nullptr; t.b[8]=W(12); // full_att
  t.w[9]=W(16); t.s[9]=W(15);   t.b[9]=W(14); // full_decomp
  t.w[10]=W(80); t.s[10]=W(78); t.b[10]=W(76); // up_ph
  t.w[11]=W(81); t.s[11]=W(79); t.b[11]=W(77);
  t.w[12]=W(74); t.s[12]=W(72); t.b[12]=W(70); // up_lh
  t.w[13]=W(75); t.s[13]=W(73); t.b[13]=W(71);
  t.w[14]=W(86); t.s[14]=W(84); t.b[14]=W(82); // up_update
  t.w[15]=W(87); t.s[15]=W(85); t.b[15]=W(83);
  t.w[16]=W(45); t.s[16]=W(43); t.b[16]=W(41); // lo_ph
  t.w[17]=W(46); t.s[17]=W(44); t.b[17]=W(42);
  t.w[18]=W(39); t.s[18]=W(37); t.b[18]=W(35); // lo_lh
  t.w[19]=W(40); t.s[19]=W(38); t.b[19]=W(36);
  t.w[20]=W(51); t.s[20]=W(49); t.b[20]=W(47); // lo_update
  t.w[21]=W(52); t.s[21]=W(50); t.b[21]=W(48);
  t.w[22]=W(95); t.s[22]=nullptr; t.b[22]=W(94); // upper_att
  t.w[23]=W(54); t.s[23]=nullptr; t.b[23]=W(53); // lower_att
  t.w[24]=W(98); t.s[24]=W(97);   t.b[24]=W(96); // upper_decomp
  t.w[25]=W(57); t.s[25]=W(56);   t.b[25]=W(55); // lower_decomp
  for (int j = 0; j < 6; ++j) {                  // node / upd (stacked [6,...])
    t.w[26+j]=W(62)+(size_t)j*32*306; t.s[26+j]=W(60)+j*32; t.b[26+j]=W(58)+j*32;
    t.w[32+j]=W(63)+(size_t)j*10*32;  t.s[32+j]=W(61)+j*10; t.b[32+j]=W(59)+j*10;
    t.w[38+j]=W(92)+(size_t)j*32*20;  t.s[38+j]=W(90)+j*32; t.b[38+j]=W(88)+j*32;
    t.w[44+j]=W(93)+(size_t)j*10*32;  t.s[44+j]=W(91)+j*10; t.b[44+j]=W(89)+j*10;
  }
  t.w[50]=W(69); t.s[50]=W(68); t.b[50]=W(67);   // p_dconv
  t.w[51]=W(34); t.s[51]=W(33); t.b[51]=W(32);   // h_dconv
  t.w[52]=W(11); t.s[52]=W(10); t.b[52]=W(9);    // f_dconv

  size_t wbytes = (size_t)woff(NL) * 2;
  wbytes = (wbytes + 255) & ~(size_t)255;
  const size_t bbytes = (size_t)boff(NL) * 4;
  if (ws_size < wbytes + bbytes) return;
  __bf16* wbuf = (__bf16*)d_ws;
  float*  bbuf = (float*)((char*)d_ws + wbytes);

  pack_weights<<<NL, 256, 0, stream>>>(t, wbuf, bbuf);

  const size_t lds = (size_t)LDS_ELEMS * 2 + 48 * 4;   // ~44.7 KB / workgroup
  gnn_main<<<(4 * S_SP) / 16, 32, lds, stream>>>(wbuf, bbuf, W(0), W(1), W(2), (float*)d_out);
}
